// ParticleEmbedding_81664508166587
// MI455X (gfx1250) — matrix operand order repaired and checked by reading, not compiled, not run on hardware
//
#include <hip/hip_runtime.h>
#include <hip/hip_bf16.h>
#include <math.h>

#define EMBED   128
#define IN_CH   7
#define EDGE_DIM 6

typedef __bf16 bf16_t;
typedef bf16_t v16bf __attribute__((ext_vector_type(16)));
typedef float  v8f   __attribute__((ext_vector_type(8)));

union AB32 { uint4 q[2]; v16bf v; };

__device__ __forceinline__ unsigned short f32_to_bf16_rne(float f) {
  unsigned int u = __float_as_uint(f);
  unsigned int r = u + 0x7fffu + ((u >> 16) & 1u);   // round-to-nearest-even
  return (unsigned short)(r >> 16);
}

__device__ __forceinline__ float gelu_exact(float v) {
  return 0.5f * v * (1.0f + erff(v * 0.7071067811865476f));
}

// ---------------------------------------------------------------------------
// Zero the message accumulator (d_out) and edge-count buffer.
__global__ __launch_bounds__(256)
void zero_kernel(float* __restrict__ out, float* __restrict__ cnt,
                 int total, int Nn) {
  int idx = blockIdx.x * 256 + threadIdx.x;
  if (idx < total) out[idx] = 0.0f;
  if (idx < Nn)    cnt[idx] = 0.0f;
}

// Convert W2 [128, 896] f32 -> bf16 (same flat layout; B_i rows are contiguous)
__global__ __launch_bounds__(256)
void cvt_bf16_kernel(const float* __restrict__ W2,
                     unsigned short* __restrict__ W2bf, int n) {
  int i = blockIdx.x * 256 + threadIdx.x;
  if (i < n) W2bf[i] = f32_to_bf16_rne(W2[i]);
}

// ---------------------------------------------------------------------------
// Main fused edge kernel: 1 block = 64 edges, 8 waves (wave w owns N-tile w).
//   msg[e,:] = sum_i x[src[e],i] * (gelu(ea[e]@W1+b1) @ W2[:, i*128:(i+1)*128])
//            + sum_i x[src[e],i] * b2[i*128:(i+1)*128]
//   atomically accumulated into out_sum[dst[e],:], counts into cnt[dst[e]].
__global__ __launch_bounds__(256)
void nnconv_edge_kernel(const float* __restrict__ x,
                        const long long* __restrict__ edge_index,  // [2,E] i64
                        const float* __restrict__ edge_attr,       // [E,6]
                        const float* __restrict__ W1,              // [6,128]
                        const float* __restrict__ b1,              // [128]
                        const unsigned short* __restrict__ W2bf,   // [128,896] bf16
                        const float* __restrict__ b2,              // [896]
                        float* __restrict__ out_sum,               // [N,128]
                        float* __restrict__ cnt,                   // [N]
                        int E) {
  __shared__ unsigned short Hs[64 * 136];  // H tile, bf16, padded row stride
  __shared__ float Xs[64 * 8];             // gathered x[src] rows (+pad)
  __shared__ int   Dsts[64];
  __shared__ float W1s[EDGE_DIM * EMBED];
  __shared__ float B1s[EMBED];

  const int t     = threadIdx.x;
  const int ebase = blockIdx.x * 64;

  // ---- stage weights + per-edge gather -----------------------------------
  for (int i = t; i < EDGE_DIM * EMBED; i += 256) W1s[i] = W1[i];
  if (t < EMBED) B1s[t] = b1[t];
  if (t < 64) {
    int ge = ebase + t;
    if (ge < E) {
      long long s = edge_index[ge];
      long long d = edge_index[(size_t)E + ge];
      Dsts[t] = (int)d;
      #pragma unroll
      for (int i = 0; i < IN_CH; ++i) Xs[t * 8 + i] = x[(size_t)s * IN_CH + i];
      Xs[t * 8 + 7] = 0.0f;
      unsafeAtomicAdd(&cnt[(int)d], 1.0f);
    } else {
      Dsts[t] = 0;
      #pragma unroll
      for (int i = 0; i < 8; ++i) Xs[t * 8 + i] = 0.0f;
    }
  }
  __syncthreads();

  // ---- H = gelu(ea @ W1 + b1) in bf16; thread -> (edge t>>2, 32 cols) -----
  {
    const int e   = t >> 2;
    const int kk0 = (t & 3) << 5;
    const int ge  = ebase + e;
    float a[EDGE_DIM];
    #pragma unroll
    for (int d = 0; d < EDGE_DIM; ++d)
      a[d] = (ge < E) ? edge_attr[(size_t)ge * EDGE_DIM + d] : 0.0f;
    #pragma unroll 8
    for (int k = 0; k < 32; ++k) {
      int kk = kk0 + k;
      float v = B1s[kk];
      #pragma unroll
      for (int d = 0; d < EDGE_DIM; ++d) v += a[d] * W1s[d * EMBED + kk];
      v = (ge < E) ? gelu_exact(v) : 0.0f;
      Hs[e * 136 + kk] = f32_to_bf16_rne(v);
    }
  }
  __syncthreads();

  // ---- WMMA GEMM: per wave, N-tile = wave id; 4 M-tiles x 7 i x 4 k-steps -
  const int lane  = t & 31;
  const int wv    = t >> 5;       // n-tile 0..7
  const int khalf = lane >> 4;    // A-fragment K-half select (ISA layout)
  const int mrow  = lane & 15;

  // per-lane base pointers; all per-(m,j,ks,i) indices become compile-time
  // immediates so DS loads need no per-access VALU address math.
  const unsigned short* hp = &Hs[mrow * 136 + (khalf << 3)];  // + m*2176 + ks*32 (+16)
  const float*          xp = &Xs[(khalf << 3) * 8];           // + m*128 + j*8 + i
  const int*            dp = &Dsts[khalf << 3];               // + m*16 + j
  const unsigned short* wp = W2bf + (size_t)(khalf << 3) * (IN_CH * EMBED) + wv * 16 + mrow;

  float acc[4][8];
  #pragma unroll
  for (int m = 0; m < 4; ++m)
    #pragma unroll
    for (int j = 0; j < 8; ++j) acc[m][j] = 0.0f;

  #pragma unroll 1
  for (int i = 0; i < IN_CH; ++i) {
    v8f pi[4];
    #pragma unroll
    for (int m = 0; m < 4; ++m)
      #pragma unroll
      for (int j = 0; j < 8; ++j) pi[m][j] = 0.0f;

    #pragma unroll
    for (int ks = 0; ks < 4; ++ks) {
      // B fragment: lane holds column N = mrow, K in ISA 16-bit table order (bf16)
      AB32 bu;
      const unsigned short* bp = wp + (size_t)ks * 32 * (IN_CH * EMBED) + i * EMBED;
      #pragma unroll
      for (int s = 0; s < 16; ++s)
        bu.v[s] = __builtin_bit_cast(bf16_t, bp[(size_t)(s < 8 ? s : s + 8) * (IN_CH * EMBED)]);
      #pragma unroll
      for (int m = 0; m < 4; ++m) {
        // A fragment from LDS: row M = mrow, K chunks per ISA 16-bit layout
        AB32 au;
        const unsigned short* ap = hp + m * 2176 + ks * 32;
        au.q[0] = *(const uint4*)(ap);
        au.q[1] = *(const uint4*)(ap + 16);
        pi[m] = __builtin_amdgcn_wmma_f32_16x16x32_bf16(
            false, au.v, false, bu.v, (short)0, pi[m], false, false);
      }
    }
    // acc += x[e,i] * P_i   (C/D layout: N = lane&15, M = j + 8*(lane>>4))
    #pragma unroll
    for (int m = 0; m < 4; ++m)
      #pragma unroll
      for (int j = 0; j < 8; ++j)
        acc[m][j] += xp[m * 128 + j * 8 + i] * pi[m][j];
  }

  // ---- epilogue: + x@b2mat, atomic scatter to out_sum[dst] ---------------
  const int ncol = wv * 16 + mrow;
  float b2v[IN_CH];
  #pragma unroll
  for (int i = 0; i < IN_CH; ++i) b2v[i] = b2[i * EMBED + ncol];

  #pragma unroll
  for (int m = 0; m < 4; ++m) {
    #pragma unroll
    for (int j = 0; j < 8; ++j) {
      int e = m * 16 + j + (khalf << 3);
      if (ebase + e < E) {
        float tb = 0.0f;
        #pragma unroll
        for (int i = 0; i < IN_CH; ++i) tb += xp[m * 128 + j * 8 + i] * b2v[i];
        unsigned off = ((unsigned)dp[m * 16 + j] << 7) + (unsigned)ncol;
        unsafeAtomicAdd(out_sum + off, acc[m][j] + tb);
      }
    }
  }
}

// ---------------------------------------------------------------------------
// out = out / max(cnt,1) + x @ root + bias
__global__ __launch_bounds__(256)
void finalize_kernel(float* __restrict__ out, const float* __restrict__ cnt,
                     const float* __restrict__ x, const float* __restrict__ root,
                     const float* __restrict__ bias, int total) {
  int idx = blockIdx.x * 256 + threadIdx.x;
  if (idx >= total) return;
  int n = idx >> 7;
  int o = idx & 127;
  float c = cnt[n];
  if (c < 1.0f) c = 1.0f;
  float r = 0.0f;
  #pragma unroll
  for (int i = 0; i < IN_CH; ++i) r += x[(size_t)n * IN_CH + i] * root[i * EMBED + o];
  out[idx] = out[idx] / c + r + bias[o];
}

// ---------------------------------------------------------------------------
extern "C" void kernel_launch(void* const* d_in, const int* in_sizes, int n_in,
                              void* d_out, int out_size, void* d_ws, size_t ws_size,
                              hipStream_t stream) {
  const float*     x    = (const float*)d_in[0];
  const long long* ei   = (const long long*)d_in[1];   // int64 [2,E]
  const float*     ea   = (const float*)d_in[2];
  const float*     W1   = (const float*)d_in[3];
  const float*     b1   = (const float*)d_in[4];
  const float*     W2   = (const float*)d_in[5];
  const float*     b2   = (const float*)d_in[6];
  const float*     root = (const float*)d_in[7];
  const float*     bias = (const float*)d_in[8];

  const int N   = in_sizes[0] / IN_CH;
  const int E   = in_sizes[1] / 2;
  const int W2n = in_sizes[5];                 // 128*896

  float* out = (float*)d_out;
  float* cnt = (float*)d_ws;
  size_t cnt_bytes = (((size_t)N * sizeof(float)) + 255) & ~(size_t)255;
  unsigned short* W2bf = (unsigned short*)((char*)d_ws + cnt_bytes);

  // 1) zero accumulators
  {
    int grid = (out_size + 255) / 256;
    zero_kernel<<<grid, 256, 0, stream>>>(out, cnt, out_size, N);
  }
  // 2) W2 -> bf16 table
  {
    int grid = (W2n + 255) / 256;
    cvt_bf16_kernel<<<grid, 256, 0, stream>>>(W2, W2bf, W2n);
  }
  // 3) fused edge MLP + per-edge matmul + scatter (WMMA)
  {
    int grid = (E + 63) / 64;
    nnconv_edge_kernel<<<grid, 256, 0, stream>>>(x, ei, ea, W1, b1, W2bf, b2,
                                                 out, cnt, E);
  }
  // 4) mean + root + bias
  {
    int grid = (out_size + 255) / 256;
    finalize_kernel<<<grid, 256, 0, stream>>>(out, cnt, x, root, bias, out_size);
  }
}
